// CapsuleLayer_56899726738051
// MI455X (gfx1250) — matrix operand order repaired and checked by reading, not compiled, not run on hardware
//
#include <hip/hip_runtime.h>
#include <math.h>

typedef __attribute__((ext_vector_type(16))) _Float16 v16h;
typedef __attribute__((ext_vector_type(8)))  _Float16 v8h;
typedef __attribute__((ext_vector_type(8)))  float    v8f;

#define BATCH   64
#define ICAPS   2048
#define DIN     8
#define NCAPS   32
#define DOUT    16
#define KSLICES 8

// workspace byte offsets (d_ws assumed >= 64 MB, 256B-aligned)
#define OFF_XH 0u          // f16 [i,b,di]            2 MB
#define OFF_WH 2097152u    // f16 [n,i,do,di]        16 MB
#define OFF_WT 18874368u   // f16 [n,i,di,do]        16 MB
#define OFF_CH 35651584u   // f16 [n,i,b]             8 MB
#define OFF_BL 44040192u   // f32 [n,i,b]            16 MB
#define OFF_SP 60817408u   // f32 [ks][b,n,do]        1 MB
#define OFF_OH 61865984u   // f16 [n,b,do]           64 KB

// ---------------- init / conversion ----------------
__global__ void k_convert_w(const float* __restrict__ W,
                            _Float16* __restrict__ Wh,
                            _Float16* __restrict__ Wt) {
  int idx = blockIdx.x * 256 + threadIdx.x;          // over N*I*DOUT*DIN = 8388608
  _Float16 hw = (_Float16)W[idx];
  Wh[idx] = hw;
  int di = idx & 7;
  int dd = (idx >> 3) & 15;
  int ni = idx >> 7;                                  // n*ICAPS + i
  Wt[(ni * DIN + di) * DOUT + dd] = hw;
}

__global__ void k_convert_x(const float* __restrict__ X,
                            _Float16* __restrict__ xh) {
  int idx = blockIdx.x * 256 + threadIdx.x;          // over B*I*DIN = 1048576
  int di = idx & 7;
  int i  = (idx >> 3) & (ICAPS - 1);
  int b  = idx >> 14;
  xh[(i * BATCH + b) * DIN + di] = (_Float16)X[idx];
}

__global__ void k_zero(float* __restrict__ bl) {
  int idx = blockIdx.x * 256 + threadIdx.x;          // over N*I*B = 4194304
  bl[idx] = 0.0f;
}

// ---------------- softmax over n (per (i,b)) ----------------
__global__ void k_softmax(const float* __restrict__ bl,
                          _Float16* __restrict__ ch) {
  int t = blockIdx.x * 256 + threadIdx.x;            // t = i*BATCH + b, 131072 total
  float v[NCAPS];
  float m = -3.0e38f;
#pragma unroll
  for (int n = 0; n < NCAPS; ++n) {
    v[n] = bl[n * (ICAPS * BATCH) + t];
    m = fmaxf(m, v[n]);
  }
  float sum = 0.0f;
#pragma unroll
  for (int n = 0; n < NCAPS; ++n) {
    v[n] = __expf(v[n] - m);
    sum += v[n];
  }
  float inv = 1.0f / sum;
#pragma unroll
  for (int n = 0; n < NCAPS; ++n)
    ch[n * (ICAPS * BATCH) + t] = (_Float16)(v[n] * inv);
}

// ---------------- s = sum_i c*h  via WMMA (M=64b x N=16do x K=16384) ----------------
__global__ void __launch_bounds__(256)
k_sgemm(const _Float16* __restrict__ xh,
        const _Float16* __restrict__ Wh,
        const _Float16* __restrict__ ch,
        float* __restrict__ sp) {
  const int n    = blockIdx.x / KSLICES;
  const int ks   = blockIdx.x % KSLICES;
  const int tid  = threadIdx.x;
  const int w    = tid >> 5;
  const int lane = tid & 31;
  const int lb   = lane & 15;
  const int h    = lane >> 4;

  v8f acc[4];
#pragma unroll
  for (int t = 0; t < 4; ++t) acc[t] = (v8f){};

  const int gw = ks * 8 + w;                         // 0..63 wave-slot within this n
#pragma unroll 1
  for (int p = 0; p < 8; ++p) {
    const int cc = gw + 64 * p;                      // chunk of 4 i's; 512 chunks total
    const int i0 = cc * 4;
    // B tile: B[k=8h+...] ii = h and 2+h: Wh[n, i0 + ii, do, di]
    v8h blo = *(const v8h*)(Wh + ((n * ICAPS + i0 + h    ) * DOUT + lb) * DIN);
    v8h bhi = *(const v8h*)(Wh + ((n * ICAPS + i0 + 2 + h) * DOUT + lb) * DIN);
    v16h Bm = __builtin_shufflevector(blo, bhi, 0,1,2,3,4,5,6,7,8,9,10,11,12,13,14,15);
#pragma unroll
    for (int t = 0; t < 4; ++t) {
      const int bg = t * 16 + lb;
      // A tile: A[m=b, k=8h+...] = c[b,n,i]*x[i,b,di],  ii = h and 2+h
      v8h alo = *(const v8h*)(xh + ((i0 + h    ) * BATCH + bg) * DIN);
      v8h ahi = *(const v8h*)(xh + ((i0 + 2 + h) * BATCH + bg) * DIN);
      _Float16 c0 = ch[(n * ICAPS + i0 + h    ) * BATCH + bg];
      _Float16 c1 = ch[(n * ICAPS + i0 + 2 + h) * BATCH + bg];
      v8h a0 = alo * c0;
      v8h a1 = ahi * c1;
      v16h Am = __builtin_shufflevector(a0, a1, 0,1,2,3,4,5,6,7,8,9,10,11,12,13,14,15);
      acc[t] = __builtin_amdgcn_wmma_f32_16x16x32_f16(
                 false, Am, false, Bm, (short)0, acc[t], false, false);
    }
  }

  // cross-wave reduction in LDS, then one deterministic partial write per block
  __shared__ float red[8 * 1024];
#pragma unroll
  for (int t = 0; t < 4; ++t)
#pragma unroll
    for (int j = 0; j < 8; ++j)
      red[w * 1024 + t * 256 + (j + 8 * h) * 16 + lb] = acc[t][j];
  __syncthreads();
  for (int e = tid; e < 1024; e += 256) {
    float sum = 0.0f;
#pragma unroll
    for (int w2 = 0; w2 < 8; ++w2) sum += red[w2 * 1024 + e];
    const int t   = e >> 8;
    const int m   = (e >> 4) & 15;
    const int col = e & 15;
    const int b   = t * 16 + m;
    sp[ks * (BATCH * NCAPS * DOUT) + (b * NCAPS + n) * DOUT + col] = sum;
  }
}

// ---------------- squash: o = s*|s|^2/((1+|s|^2)sqrt(|s|^2+eps)) ----------------
__global__ void k_squash(const float* __restrict__ sp,
                         float* __restrict__ out,
                         _Float16* __restrict__ oh) {
  int t = blockIdx.x * 256 + threadIdx.x;            // t = b*NCAPS + n, 2048 total
  const int b = t >> 5;
  const int n = t & 31;
  float v[DOUT];
  float s2 = 0.0f;
#pragma unroll
  for (int k = 0; k < DOUT; ++k) {
    float a = 0.0f;
#pragma unroll
    for (int ks = 0; ks < KSLICES; ++ks)
      a += sp[ks * (BATCH * NCAPS * DOUT) + t * DOUT + k];
    v[k] = a;
    s2 += a * a;
  }
  const float scale = s2 / (1.0f + s2) / sqrtf(s2 + 1e-7f);
#pragma unroll
  for (int k = 0; k < DOUT; ++k) {
    float ov = scale * v[k];
    out[t * DOUT + k] = ov;                          // [b,n,do] == reference layout
    oh[(n * BATCH + b) * DOUT + k] = (_Float16)ov;
  }
}

// ---------------- b += sum_do o*h  via WMMA (U[b,(ii,di)] = o @ W^T) ----------------
__global__ void __launch_bounds__(256)
k_update(const _Float16* __restrict__ xh,
         const _Float16* __restrict__ Wt,
         const _Float16* __restrict__ oh,
         float* __restrict__ bl) {
  const int n    = blockIdx.x >> 4;
  const int blk  = blockIdx.x & 15;
  const int tid  = threadIdx.x;
  const int w    = tid >> 5;
  const int lane = tid & 31;
  const int lb   = lane & 15;
  const int h    = lane >> 4;
  const int ii   = lb >> 3;
  const int di   = lb & 7;

  const v8h z8 = (v8h){};
  v16h av[4];
#pragma unroll
  for (int t = 0; t < 4; ++t) {
    // A[m=b, k=do<16]=o[b,n,do]; k>=16 zero-padded
    v8h olo = *(const v8h*)(oh + (n * BATCH + t * 16 + lb) * DOUT + 8 * h);
    av[t] = __builtin_shufflevector(olo, z8, 0,1,2,3,4,5,6,7,8,9,10,11,12,13,14,15);
  }

#pragma unroll 1
  for (int q = 0; q < 8; ++q) {
    const int ip = blk * 64 + w * 8 + q;             // i-pair index, 1024 per n
    const int i0 = 2 * ip;
    // B[k=do=8h+s (s<8), nn=(ii,di)] = Wt[n, i0+ii, di, do]; k>=16 zero-padded
    v8h blo = *(const v8h*)(Wt + ((n * ICAPS + i0 + ii) * DIN + di) * DOUT + 8 * h);
    v16h Bm = __builtin_shufflevector(blo, z8, 0,1,2,3,4,5,6,7,8,9,10,11,12,13,14,15);
#pragma unroll
    for (int t = 0; t < 4; ++t) {
      v8f zc = (v8f){};
      v8f d = __builtin_amdgcn_wmma_f32_16x16x32_f16(
                false, av[t], false, Bm, (short)0, zc, false, false);
#pragma unroll
      for (int j = 0; j < 8; ++j) {
        const int b = t * 16 + j + 8 * h;
        float xv  = (float)xh[((i0 + ii) * BATCH + b) * DIN + di];
        float val = d[j] * xv;
        val += __shfl_xor(val, 1);
        val += __shfl_xor(val, 2);
        val += __shfl_xor(val, 4);
        if ((lane & 7) == 0)                          // unique (n,i,b) writer
          bl[(n * ICAPS + i0 + ii) * BATCH + b] += val;
      }
    }
  }
}

// ---------------- host launch ----------------
extern "C" void kernel_launch(void* const* d_in, const int* in_sizes, int n_in,
                              void* d_out, int out_size, void* d_ws, size_t ws_size,
                              hipStream_t stream) {
  const float* X = (const float*)d_in[0];            // [64,2048,8]
  const float* W = (const float*)d_in[1];            // [32,2048,16,8]
  float* out = (float*)d_out;                        // [64,32,16]

  char* ws = (char*)d_ws;
  _Float16* xh = (_Float16*)(ws + OFF_XH);
  _Float16* Wh = (_Float16*)(ws + OFF_WH);
  _Float16* Wt = (_Float16*)(ws + OFF_WT);
  _Float16* ch = (_Float16*)(ws + OFF_CH);
  float*    bl = (float*)   (ws + OFF_BL);
  float*    sp = (float*)   (ws + OFF_SP);
  _Float16* oh = (_Float16*)(ws + OFF_OH);

  k_convert_w<<<32768, 256, 0, stream>>>(W, Wh, Wt);
  k_convert_x<<<4096, 256, 0, stream>>>(X, xh);
  k_zero<<<16384, 256, 0, stream>>>(bl);

  for (int r = 0; r < 3; ++r) {
    k_softmax<<<512, 256, 0, stream>>>(bl, ch);
    k_sgemm<<<NCAPS * KSLICES, 256, 0, stream>>>(xh, Wh, ch, sp);
    k_squash<<<8, 256, 0, stream>>>(sp, out, oh);
    if (r < 2)
      k_update<<<NCAPS * 16, 256, 0, stream>>>(xh, Wt, oh, bl);
  }
}
